// Text_Encoder_11141145166141
// MI455X (gfx1250) — compile-verified, matrix operand order checked by reading, not run on hardware
//
#include <hip/hip_runtime.h>
#include <hip/hip_bf16.h>

// Shapes from the reference
#define BATCH 1024
#define TSEQ  100
#define DIM   128
#define HID   64
#define GATES 256   // 4*H

typedef __attribute__((ext_vector_type(8)))  __bf16 v8bf;
typedef __attribute__((ext_vector_type(16))) __bf16 v16bf;
typedef __attribute__((ext_vector_type(8)))  float  v8f;

__device__ inline v16bf ldfrag(const __bf16* lo, const __bf16* hi) {
    v8bf a = *(const v8bf*)lo;
    v8bf b = *(const v8bf*)hi;
    v16bf r;
#pragma unroll
    for (int i = 0; i < 8; ++i) { r[i] = a[i]; r[i + 8] = b[i]; }
    return r;
}

__device__ inline v8f wmma_bf16(v16bf a, v16bf b, v8f c) {
    return __builtin_amdgcn_wmma_f32_16x16x32_bf16(
        /*neg_a=*/false, a, /*neg_b=*/false, b,
        /*c_mod=*/(short)0, c, /*reuse_a=*/false, /*reuse_b=*/false);
}

__device__ inline float sigm(float x) { return 1.0f / (1.0f + __expf(-x)); }

// ---------------------------------------------------------------------------
// Kernel 0: cast weights to bf16 into workspace, pre-sum biases.
// ws layout: [0)      wih_bf  256*128 bf16  (65536 B)
//            [65536)  whh_bf  256*64  bf16  (32768 B)
//            [98304)  bias    256 f32       (1024 B)
// ---------------------------------------------------------------------------
__global__ void convert_weights(const float* __restrict__ W_ih,
                                const float* __restrict__ W_hh,
                                const float* __restrict__ b_ih,
                                const float* __restrict__ b_hh,
                                __bf16* __restrict__ wih_bf,
                                __bf16* __restrict__ whh_bf,
                                float* __restrict__ biasws) {
    int i = blockIdx.x * blockDim.x + threadIdx.x;
    if (i < GATES * DIM) {
        wih_bf[i] = (__bf16)W_ih[i];
    } else if (i < GATES * DIM + GATES * HID) {
        int j = i - GATES * DIM;
        whh_bf[j] = (__bf16)W_hh[j];
    } else if (i < GATES * DIM + GATES * HID + GATES) {
        int j = i - GATES * DIM - GATES * HID;
        biasws[j] = b_ih[j] + b_hh[j];
    }
}

// ---------------------------------------------------------------------------
// Kernel 1: fused gather + LSTM recurrence + MLP head.
// Grid: BATCH/16 workgroups, 256 threads (8 wave32). Each WG owns 16 batch
// rows for all T steps. All weight fragments are register-resident; the
// embedding gather for step t+1 is double-buffered and issued at the top of
// step t so its HBM latency overlaps the WMMA + gate phases.
// ---------------------------------------------------------------------------
__global__ __launch_bounds__(256) void lstm_fused(
    const int* __restrict__ id_batch,
    const int* __restrict__ p_content,
    const float* __restrict__ word_embed,
    const __bf16* __restrict__ wih,     // [256][128] bf16
    const __bf16* __restrict__ whh,     // [256][64]  bf16
    const float* __restrict__ bias,     // [256] (b_ih + b_hh)
    const float* __restrict__ W_fchl,   // [32][64]
    const float* __restrict__ b_fchl,   // [32]
    const float* __restrict__ W_fcol,   // [5][32]
    const float* __restrict__ b_fcol,   // [5]
    float* __restrict__ out)            // [B][5]
{
    __shared__ __align__(16) __bf16 sX[2][16][DIM];     //  8 KB (double buffer)
    __shared__ __align__(16) __bf16 sHb[16][HID];       //  2 KB
    __shared__ float  sHf[16][HID];                     //  4 KB
    __shared__ float  sC[16][HID];                      //  4 KB
    __shared__ float  sG[16][GATES];                    // 16 KB
    __shared__ int    sTokAll[16][TSEQ];                // 6.4 KB

    const int tid  = threadIdx.x;
    const int lane = tid & 31;
    const int wave = tid >> 5;
    const int blk  = blockIdx.x;

    // ---- preload ALL tokens for this block's 16 docs (removes the per-step
    //      token->embedding dependency chain entirely)
    for (int i = tid; i < 16 * TSEQ; i += 256) {
        int r  = i / TSEQ;
        int tt = i - r * TSEQ;
        int doc = id_batch[blk * 16 + r];
        sTokAll[r][tt] = p_content[(size_t)doc * TSEQ + tt];
    }
    // ---- init h, c
    for (int i = tid; i < 16 * HID; i += 256) {
        ((float*)sHf)[i]  = 0.0f;
        ((float*)sC)[i]   = 0.0f;
        ((__bf16*)sHb)[i] = (__bf16)0.0f;
    }
    __syncthreads();

    // WMMA fragment addressing (wave32, 16-bit A/B layout):
    //   lanes 0-15 -> K 0-7 / 16-23 halves, lanes 16-31 -> K 8-15 / 24-31.
    const int row = lane & 15;
    const int kh  = (lane >> 4) * 8;
    const int nc0 = (wave * 2)     * 16 + row;  // gate column, N-tile 0
    const int nc1 = (wave * 2 + 1) * 16 + row;  // gate column, N-tile 1
    const float bias0 = bias[nc0];
    const float bias1 = bias[nc1];

    // ---- hoist ALL weight B-fragments to registers (loop-invariant):
    //      W_ih: 2 tiles x 4 K-chunks, W_hh: 2 tiles x 2 K-chunks.
    v16bf bx0[4], bx1[4], bh0[2], bh1[2];
#pragma unroll
    for (int k = 0; k < 4; ++k) {
        const __bf16* p0 = wih + (size_t)nc0 * DIM + k * 32 + kh;
        const __bf16* p1 = wih + (size_t)nc1 * DIM + k * 32 + kh;
        bx0[k] = ldfrag(p0, p0 + 16);
        bx1[k] = ldfrag(p1, p1 + 16);
    }
#pragma unroll
    for (int k = 0; k < 2; ++k) {
        const __bf16* p0 = whh + (size_t)nc0 * HID + k * 32 + kh;
        const __bf16* p1 = whh + (size_t)nc1 * HID + k * 32 + kh;
        bh0[k] = ldfrag(p0, p0 + 16);
        bh1[k] = ldfrag(p1, p1 + 16);
    }

    // ---- prologue: gather x[0] into buffer 0
    {
        int r  = tid >> 4;
        int dc = (tid & 15) * 8;
        const float* e = word_embed + (size_t)sTokAll[r][0] * DIM + dc;
#pragma unroll
        for (int j = 0; j < 8; ++j) sX[0][r][dc + j] = (__bf16)e[j];
    }
    __syncthreads();

    for (int t = 0; t < TSEQ; ++t) {
        const int cur = t & 1;
        const int nxt = cur ^ 1;

        // ---- issue gather for step t+1 early: HBM latency overlaps WMMAs
        if (t + 1 < TSEQ) {
            int r  = tid >> 4;
            int dc = (tid & 15) * 8;
            const float* e = word_embed + (size_t)sTokAll[r][t + 1] * DIM + dc;
#pragma unroll
            for (int j = 0; j < 8; ++j) sX[nxt][r][dc + j] = (__bf16)e[j];
        }

        // ---- A fragments (shared by both N-tiles of this wave)
        v16bf ax[4];
#pragma unroll
        for (int k = 0; k < 4; ++k)
            ax[k] = ldfrag(&sX[cur][row][k * 32 + kh],
                           &sX[cur][row][k * 32 + kh + 16]);
        v16bf ah[2];
#pragma unroll
        for (int k = 0; k < 2; ++k)
            ah[k] = ldfrag(&sHb[row][k * 32 + kh], &sHb[row][k * 32 + kh + 16]);

        // ---- gates = x @ W_ih^T + h @ W_hh^T   (12 WMMAs per wave)
        v8f acc0 = {};
        v8f acc1 = {};
#pragma unroll
        for (int k = 0; k < 4; ++k) {
            acc0 = wmma_bf16(ax[k], bx0[k], acc0);
            acc1 = wmma_bf16(ax[k], bx1[k], acc1);
        }
#pragma unroll
        for (int k = 0; k < 2; ++k) {
            acc0 = wmma_bf16(ah[k], bh0[k], acc0);
            acc1 = wmma_bf16(ah[k], bh1[k], acc1);
        }

        // ---- scatter accumulators to gate LDS (C layout: VGPR r -> M, lane -> N)
#pragma unroll
        for (int r8 = 0; r8 < 8; ++r8) {
            int m = (lane >> 4) * 8 + r8;
            sG[m][nc0] = acc0[r8] + bias0;
            sG[m][nc1] = acc1[r8] + bias1;
        }
        __syncthreads();

        // ---- elementwise LSTM cell update (i,f,g,o gate order)
#pragma unroll
        for (int j = 0; j < 4; ++j) {
            int cell = tid + j * 256;          // 16*64 cells / 256 threads
            int r  = cell >> 6;
            int hh = cell & 63;
            float i_ = sigm(sG[r][hh]);
            float f_ = sigm(sG[r][HID + hh]);
            float g_ = tanhf(sG[r][2 * HID + hh]);
            float o_ = sigm(sG[r][3 * HID + hh]);
            float c  = f_ * sC[r][hh] + i_ * g_;
            float h  = o_ * tanhf(c);
            sC[r][hh]  = c;
            sHf[r][hh] = h;
            sHb[r][hh] = (__bf16)h;
        }
        __syncthreads();   // sHb + sX[nxt] visible for next step
    }

    // ---- MLP head: relu -> sigmoid(64->32) -> 32->5 (scalar, 16 rows)
    if (tid < 16) {
        float hrelu[HID];
#pragma unroll
        for (int k = 0; k < HID; ++k) hrelu[k] = fmaxf(sHf[tid][k], 0.0f);
        float hid[32];
        for (int j = 0; j < 32; ++j) {
            float s = b_fchl[j];
            for (int k = 0; k < HID; ++k) s += W_fchl[j * HID + k] * hrelu[k];
            hid[j] = sigm(s);
        }
        for (int o = 0; o < 5; ++o) {
            float s = b_fcol[o];
            for (int j = 0; j < 32; ++j) s += W_fcol[o * 32 + j] * hid[j];
            out[(size_t)(blk * 16 + tid) * 5 + o] = s;
        }
    }
}

// ---------------------------------------------------------------------------
extern "C" void kernel_launch(void* const* d_in, const int* in_sizes, int n_in,
                              void* d_out, int out_size, void* d_ws, size_t ws_size,
                              hipStream_t stream) {
    const int*   id_batch  = (const int*)d_in[0];
    const int*   p_content = (const int*)d_in[1];
    const float* word_embed= (const float*)d_in[2];
    const float* W_ih      = (const float*)d_in[3];
    const float* W_hh      = (const float*)d_in[4];
    const float* b_ih      = (const float*)d_in[5];
    const float* b_hh      = (const float*)d_in[6];
    const float* W_fchl    = (const float*)d_in[7];
    const float* b_fchl    = (const float*)d_in[8];
    const float* W_fcol    = (const float*)d_in[9];
    const float* b_fcol    = (const float*)d_in[10];
    float* out = (float*)d_out;

    __bf16* wih_bf = (__bf16*)d_ws;                         // 65536 B
    __bf16* whh_bf = wih_bf + GATES * DIM;                  // 32768 B
    float*  biasws = (float*)((char*)d_ws + 65536 + 32768); // 1024 B

    int conv_n = GATES * DIM + GATES * HID + GATES;         // 49408
    convert_weights<<<(conv_n + 255) / 256, 256, 0, stream>>>(
        W_ih, W_hh, b_ih, b_hh, wih_bf, whh_bf, biasws);

    lstm_fused<<<BATCH / 16, 256, 0, stream>>>(
        id_batch, p_content, word_embed, wih_bf, whh_bf, biasws,
        W_fchl, b_fchl, W_fcol, b_fcol, out);
}
